// BasicBlockBase_85409719648346
// MI455X (gfx1250) — matrix operand order repaired and checked by reading, not compiled, not run on hardware
//
#include <hip/hip_runtime.h>
#include <hip/hip_bf16.h>

typedef __attribute__((ext_vector_type(16))) _Float16 v16h;
typedef __attribute__((ext_vector_type(8)))  _Float16 v8h;
typedef __attribute__((ext_vector_type(8)))  float    v8f;
typedef __attribute__((ext_vector_type(4)))  int      v4i;

union Frag { v16h v; v8h h[2]; };

#if __has_builtin(__builtin_amdgcn_global_load_async_to_lds_b128)
#define ASYNC_LDS 1
#else
#define ASYNC_LDS 0
#endif

__device__ __forceinline__ void copy16_to_lds(const _Float16* g, _Float16* l) {
#if ASYNC_LDS
    __builtin_amdgcn_global_load_async_to_lds_b128(
        (__attribute__((address_space(1))) v4i*)(v4i*)g,
        (__attribute__((address_space(3))) v4i*)(v4i*)l, 0, 0);
#else
    *(v8h*)l = *(const v8h*)g;
#endif
}

__device__ __forceinline__ void wait_async_lds() {
#if ASYNC_LDS
#if __has_builtin(__builtin_amdgcn_s_wait_asynccnt)
    __builtin_amdgcn_s_wait_asynccnt(0);
#else
    asm volatile("s_wait_asynccnt 0x0" ::: "memory");
#endif
#endif
}

// ---------------------------------------------------------------------------
// Init: Fh[:,0:64] = f16(x); Facc[N,192]=0; X4[N,64]=0; stats[512]=0
// ---------------------------------------------------------------------------
__global__ __launch_bounds__(256)
void init_kernel(const float* __restrict__ x, _Float16* __restrict__ Fh,
                 float* __restrict__ Facc, float* __restrict__ X4,
                 float* __restrict__ stats, long N) {
    long segA = N * 64, segB = segA + N * 192, segC = segB + N * 64;
    long total = segC + 512;
    long stride = (long)gridDim.x * blockDim.x;
    for (long i = (long)blockIdx.x * blockDim.x + threadIdx.x; i < total; i += stride) {
        if (i < segA) {
            long r = i >> 6; int c = (int)(i & 63);
            Fh[r * 256 + c] = (_Float16)x[i];
        } else if (i < segB) {
            Facc[i - segA] = 0.0f;
        } else if (i < segC) {
            X4[i - segB] = 0.0f;
        } else {
            stats[i - segC] = 0.0f;
        }
    }
}

// ---------------------------------------------------------------------------
// Weight prep: W [27][CIN][64] f32  ->  Wh [27][64][CIN] f16 (transposed)
// ---------------------------------------------------------------------------
__global__ __launch_bounds__(256)
void prep_w(const float* __restrict__ W, _Float16* __restrict__ Wh, int CIN) {
    int total = 27 * CIN * 64;
    int stride = gridDim.x * blockDim.x;
    for (int i = blockIdx.x * blockDim.x + threadIdx.x; i < total; i += stride) {
        int k = i / (CIN * 64);
        int rem = i % (CIN * 64);
        int ci = rem >> 6, n = rem & 63;
        Wh[((long)k * 64 + n) * CIN + ci] = (_Float16)W[i];
    }
}

// ---------------------------------------------------------------------------
// Column-block snapshot: Facc f32 block -> Fh f16 block (64 cols)
// ---------------------------------------------------------------------------
__global__ __launch_bounds__(256)
void cvt_cols(const float* __restrict__ src, int sstride,
              _Float16* __restrict__ dst, int dstride, long N) {
    long total = N * 64;
    long stride = (long)gridDim.x * blockDim.x;
    for (long i = (long)blockIdx.x * blockDim.x + threadIdx.x; i < total; i += stride) {
        long r = i >> 6; int c = (int)(i & 63);
        dst[r * dstride + c] = (_Float16)src[r * sstride + c];
    }
}

// ---------------------------------------------------------------------------
// Sparse conv tile kernel: async-gather 32 f16 rows + f16 W[k]^T tile to LDS,
// 8 waves x WMMA 16x16 tiles, scatter-add f32 atomics.
//   Fh:  [N,256] f16 gather buffer (conv inputs are cols 0:CIN)
//   Wh:  [27][64][CIN] f16 transposed weights
// ---------------------------------------------------------------------------
template <int CIN>
__global__ __launch_bounds__(256)
void sconv_wmma(const _Float16* __restrict__ Fh,
                const _Float16* __restrict__ Wh,
                const int* __restrict__ in_map, const int* __restrict__ out_map,
                float* __restrict__ out, int ostride, int M) {
    constexpr int KP  = CIN + 16;                // padded K stride (halves)
    constexpr int CH8 = CIN / 8;                 // 16B chunks per row
    __shared__ alignas(32) _Float16 sA[32][KP];  // gathered rows  [m][k]
    __shared__ alignas(32) _Float16 sB[64][KP];  // W[k]^T         [n][k]
    __shared__ int sIn[32];
    __shared__ int sOut[32];

    const int tid = threadIdx.x;
    const int ko  = blockIdx.y;                  // kernel offset 0..26
    const int m0  = blockIdx.x * 32;

    if (tid < 32) {
        int m = m0 + tid;
        sIn[tid]  = (m < M) ? in_map[(long)ko * M + m] : 0;
        sOut[tid] = (m < M) ? out_map[(long)ko * M + m] : -1;
    }
    __syncthreads();

    // A tile: 32 rows x CIN halves, gathered via async global->LDS b128
    for (int idx = tid; idx < 32 * CH8; idx += 256) {
        int r = idx / CH8, c8 = (idx % CH8) * 8;
        copy16_to_lds(Fh + (long)sIn[r] * 256 + c8, &sA[r][c8]);
    }
    // B tile: W[ko]^T, 64 rows x CIN halves, contiguous async copy
    const _Float16* Wk = Wh + (long)ko * 64 * CIN;
    for (int idx = tid; idx < 64 * CH8; idx += 256) {
        int n = idx / CH8, c8 = (idx % CH8) * 8;
        copy16_to_lds(Wk + (long)n * CIN + c8, &sB[n][c8]);
    }
    wait_async_lds();
    __syncthreads();

    const int wave = tid >> 5, lane = tid & 31;
    const int tm   = (wave >> 2) * 16;           // row-tile base: 0 / 16
    const int tn   = (wave & 3) * 16;            // col-tile base: 0/16/32/48
    const int lmod = lane & 15, lhi = lane >> 4;

    // A frag (16x32 MxK): lane<16 holds K {0..7,16..23}; lane>=16 {8..15,24..31}
    // B^T frag: same K order as A (lane<16: K {0..7,16..23}; lane>=16: {8..15,24..31})
    const _Float16* arow = &sA[tm + lmod][0];
    const _Float16* brow = &sB[tn + lmod][0];

    v8f acc = {};
#pragma unroll
    for (int kk = 0; kk < CIN; kk += 32) {
        Frag a, b;
        const int aoff = kk + lhi * 8;
        a.h[0] = *(const v8h*)(arow + aoff);
        a.h[1] = *(const v8h*)(arow + aoff + 16);
        const int boff = kk + lhi * 8;
        b.h[0] = *(const v8h*)(brow + boff);
        b.h[1] = *(const v8h*)(brow + boff + 16);
        acc = __builtin_amdgcn_wmma_f32_16x16x32_f16(
            false, a.v, false, b.v, (short)0, acc, false, false);
    }

    // Scatter-add: C/D layout -> VGPR r: lanes 0-15 (M=r,N=lane),
    // lanes 16-31 (M=r+8, N=lane-16)
#pragma unroll
    for (int r = 0; r < 8; ++r) {
        int mloc = tm + r + lhi * 8;
        int orow = sOut[mloc];
        if (orow >= 0)
            unsafeAtomicAdd(&out[(long)orow * ostride + tn + lmod], acc[r]);
    }
}

// ---------------------------------------------------------------------------
// BN statistics + apply. Output cols 0:192 = Facc (x1|x2|x3), 192:256 = X4.
// ---------------------------------------------------------------------------
__global__ __launch_bounds__(256)
void bn_stats(const float* __restrict__ Facc, const float* __restrict__ X4,
              float* __restrict__ stats, int N) {
    int c = threadIdx.x;                 // 256 columns
    float s = 0.0f, ss = 0.0f;
    for (int r = blockIdx.x; r < N; r += gridDim.x) {
        float v = (c < 192) ? Facc[(long)r * 192 + c]
                            : X4[(long)r * 64 + (c - 192)];
        s += v; ss += v * v;
    }
    unsafeAtomicAdd(&stats[c], s);
    unsafeAtomicAdd(&stats[256 + c], ss);
}

__global__ __launch_bounds__(256)
void bn_apply(const float* __restrict__ Facc, const float* __restrict__ X4,
              const float* __restrict__ stats,
              const float* __restrict__ gamma, const float* __restrict__ beta,
              float* __restrict__ out, long N, float invN) {
    long total = N * 256;
    long stride = (long)gridDim.x * blockDim.x;
    for (long i = (long)blockIdx.x * blockDim.x + threadIdx.x; i < total; i += stride) {
        int  c = (int)(i & 255);
        long r = i >> 8;
        float v = (c < 192) ? Facc[r * 192 + c] : X4[r * 64 + (c - 192)];
        float mean = stats[c] * invN;
        float var  = stats[256 + c] * invN - mean * mean;
        out[i] = (v - mean) * rsqrtf(var + 1e-5f) * gamma[c] + beta[c];
    }
}

// ---------------------------------------------------------------------------
// kernel_launch
// ---------------------------------------------------------------------------
extern "C" void kernel_launch(void* const* d_in, const int* in_sizes, int n_in,
                              void* d_out, int out_size, void* d_ws, size_t ws_size,
                              hipStream_t stream) {
    const float* x     = (const float*)d_in[0];
    const float* W1    = (const float*)d_in[1];
    const float* W2    = (const float*)d_in[2];
    const float* W3    = (const float*)d_in[3];
    const float* W4    = (const float*)d_in[4];
    const float* gamma = (const float*)d_in[5];
    const float* beta  = (const float*)d_in[6];
    const int* in_map  = (const int*)d_in[7];
    const int* out_map = (const int*)d_in[8];

    const long N = in_sizes[0] / 64;     // 120000
    const int  M = in_sizes[7] / 27;     // 60000
    const int  K = 27;

    _Float16* Fh   = (_Float16*)d_ws;            // [N,256] f16 gather buffer
    float*    Facc = (float*)(Fh + (size_t)N * 256);   // [N,192] f32: x1|x2|x3
    float*    X4   = Facc + (size_t)N * 192;           // [N,64]  f32
    float*    stats = X4 + (size_t)N * 64;             // [512]
    _Float16* Wh1  = (_Float16*)(stats + 512);         // [27][64][64]
    _Float16* Wh2  = Wh1 + (size_t)27 * 64 * 64;       // [27][64][128]
    _Float16* Wh3  = Wh2 + (size_t)27 * 64 * 128;      // [27][64][192]
    _Float16* Wh4  = Wh3 + (size_t)27 * 64 * 192;      // [27][64][256]

    init_kernel<<<2048, 256, 0, stream>>>(x, Fh, Facc, X4, stats, N);
    prep_w<<<256, 256, 0, stream>>>(W1, Wh1, 64);
    prep_w<<<256, 256, 0, stream>>>(W2, Wh2, 128);
    prep_w<<<256, 256, 0, stream>>>(W3, Wh3, 192);
    prep_w<<<256, 256, 0, stream>>>(W4, Wh4, 256);

    dim3 grid((M + 31) / 32, K);
    dim3 blk(256);
    // conv1: reads Fh[:,0:64],  accumulates into Facc[:,0:64] (= x1)
    sconv_wmma<64 ><<<grid, blk, 0, stream>>>(Fh, Wh1, in_map, out_map, Facc,       192, M);
    cvt_cols<<<2048, 256, 0, stream>>>(Facc,       192, Fh + 64,  256, N);
    // conv2: reads Fh[:,0:128], accumulates into Facc[:,64:128] (= x2)
    sconv_wmma<128><<<grid, blk, 0, stream>>>(Fh, Wh2, in_map, out_map, Facc + 64,  192, M);
    cvt_cols<<<2048, 256, 0, stream>>>(Facc + 64,  192, Fh + 128, 256, N);
    // conv3: reads Fh[:,0:192], accumulates into Facc[:,128:192] (= x3)
    sconv_wmma<192><<<grid, blk, 0, stream>>>(Fh, Wh3, in_map, out_map, Facc + 128, 192, M);
    cvt_cols<<<2048, 256, 0, stream>>>(Facc + 128, 192, Fh + 192, 256, N);
    // conv4: reads Fh[:,0:256], accumulates into X4
    sconv_wmma<256><<<grid, blk, 0, stream>>>(Fh, Wh4, in_map, out_map, X4, 64, M);

    bn_stats<<<256, 256, 0, stream>>>(Facc, X4, stats, (int)N);
    bn_apply<<<2048, 256, 0, stream>>>(Facc, X4, stats, gamma, beta,
                                       (float*)d_out, N, 1.0f / (float)N);
}
